// Decoder_1271310320240
// MI455X (gfx1250) — matrix operand order repaired and checked by reading, not compiled, not run on hardware
//
#include <hip/hip_runtime.h>
#include <stdint.h>

// ---------------- problem constants ----------------
#define DM    256          // d_model
#define NL    3            // layers
#define BATCH 1024
#define TAU   96
#define GATES 1024         // 4*DM
#define KTOT  512          // [x,h] concat

// ---------------- tiling ----------------
#define TM 128             // batch rows per block
#define TN 64              // gate cols per block (g'-space)
#define TK 32              // k per stage
#define NSTAGE (KTOT / TK) // 16
#define XSTRIDE 40         // bf16 elems per LDS row (80B, rows 16B aligned)
#define WSTRIDE 40
#define GSTRIDE 68         // f32 elems per LDS gates row (272B)

#define XBYTES (TM * XSTRIDE * 2)   // 10240
#define WBYTES (TN * WSTRIDE * 2)   // 5120
#define GBYTES (TM * GSTRIDE * 4)   // 34816
#define SMEM_BYTES 34816            // max(2*XBYTES + 2*WBYTES = 30720, GBYTES)

typedef __attribute__((ext_vector_type(16))) __bf16 v16bf;
typedef __attribute__((ext_vector_type(8)))  float  v8f;
typedef __attribute__((ext_vector_type(4)))  int    v4i;

typedef __attribute__((address_space(1))) v4i as1_v4i;
typedef __attribute__((address_space(3))) v4i as3_v4i;

#if __has_builtin(__builtin_amdgcn_global_load_async_to_lds_b128)
#define HAVE_ASYNC_LDS 1
#else
#define HAVE_ASYNC_LDS 0
#endif

__device__ __forceinline__ void copy16_to_lds(const uint16_t* g, uint16_t* l) {
#if HAVE_ASYNC_LDS
    __builtin_amdgcn_global_load_async_to_lds_b128(
        (as1_v4i*)(uintptr_t)g, (as3_v4i*)(uint32_t)(uintptr_t)l, 0, 0);
#else
    *(uint4*)l = *(const uint4*)g;
#endif
}

__device__ __forceinline__ void wait_async0() {
#if HAVE_ASYNC_LDS
#if __has_builtin(__builtin_amdgcn_s_wait_asynccnt)
    __builtin_amdgcn_s_wait_asynccnt(0);
#else
    asm volatile("s_wait_asynccnt 0x0" ::: "memory");
#endif
#endif
}

__device__ __forceinline__ uint16_t f2bf(float x) {
    uint32_t u = __float_as_uint(x);
    uint32_t r = u + 0x7FFFu + ((u >> 16) & 1u);   // round-to-nearest-even
    return (uint16_t)(r >> 16);
}
__device__ __forceinline__ float sigm(float x) {
    return 1.0f / (1.0f + __expf(-x));
}

// ---------------- prep kernels (run once per call) ----------------
// Wout[l][g'][k], g' = 4*d + gate, k<256 -> W_ih, else W_hh. bf16 bits.
__global__ __launch_bounds__(256) void prep_weights(
    const float* __restrict__ W_ih, const float* __restrict__ W_hh,
    uint16_t* __restrict__ Wout)
{
    size_t i = (size_t)blockIdx.x * 256 + threadIdx.x;   // < NL*GATES*KTOT
    int l  = (int)(i / ((size_t)GATES * KTOT));
    int r  = (int)(i % ((size_t)GATES * KTOT));
    int gp = r / KTOT;
    int k  = r % KTOT;
    int d    = gp >> 2;
    int gate = gp & 3;
    int g    = gate * DM + d;                 // original PyTorch gate row
    float v = (k < DM) ? W_ih[((size_t)l * GATES + g) * DM + k]
                       : W_hh[((size_t)l * GATES + g) * DM + (k - DM)];
    Wout[i] = f2bf(v);
}

__global__ __launch_bounds__(256) void prep_bias(
    const float* __restrict__ b_ih, const float* __restrict__ b_hh,
    float* __restrict__ bout)
{
    int i = blockIdx.x * 256 + threadIdx.x;   // < NL*GATES
    int l  = i / GATES;
    int gp = i % GATES;
    int d    = gp >> 2;
    int gate = gp & 3;
    int g    = gate * DM + d;
    bout[i] = b_ih[l * GATES + g] + b_hh[l * GATES + g];
}

__global__ __launch_bounds__(256) void prep_state(
    const float* __restrict__ hidden, const float* __restrict__ cell,
    uint16_t* __restrict__ h0, float* __restrict__ cws)
{
    size_t i = (size_t)blockIdx.x * 256 + threadIdx.x;   // < NL*BATCH*DM
    h0[i]  = f2bf(hidden[i]);
    cws[i] = cell[i];
}

// ---------------- fused LSTM cell: GEMM (WMMA bf16) + gate nonlinearity ----
// gates[B, G'] = [X|H] * Wl^T + bl ; then c,h update on the block's d-slice.
__global__ __launch_bounds__(256) void lstm_cell_kernel(
    const uint16_t* __restrict__ Wl,    // [GATES][KTOT] bf16, g'-permuted
    const float*    __restrict__ bl,    // [GATES] g'-permuted
    const uint16_t* __restrict__ Xsrc,  // [BATCH][DM] bf16
    const uint16_t* __restrict__ Hsrc,  // [BATCH][DM] bf16
    uint16_t*       __restrict__ Hdst,  // [BATCH][DM] bf16
    float*          __restrict__ Cst,   // [BATCH][DM] f32 (in/out)
    float*          __restrict__ Out,   // [BATCH][TAU][DM] f32 or null
    int t)
{
    // one byte pool: [Xs0|Xs1|Ws0|Ws1] during GEMM, reused as Gs afterwards
    __shared__ __align__(16) uint8_t smem[SMEM_BYTES];
    uint16_t* const XsB[2] = { (uint16_t*)smem,
                               (uint16_t*)(smem + XBYTES) };
    uint16_t* const WsB[2] = { (uint16_t*)(smem + 2 * XBYTES),
                               (uint16_t*)(smem + 2 * XBYTES + WBYTES) };
    float* const Gs = (float*)smem;

    const int tid  = threadIdx.x;
    const int lane = tid & 31;
    const int wv   = tid >> 5;       // 0..7 (wave32)
    const int wr   = wv >> 1;        // 0..3 : 32-row group
    const int wc   = wv & 1;         // 0..1 : 32-col group
    const int hf   = lane >> 4;      // lane half
    const int mn   = lane & 15;      // M (for A) / N (for B,C)
    const int r0   = blockIdx.y * TM;
    const int c0   = blockIdx.x * TN;
    const int ch   = tid & 3;        // 16B chunk within a 32-elem row

    v8f acc[2][2];
#pragma unroll
    for (int mt = 0; mt < 2; ++mt)
#pragma unroll
        for (int nt = 0; nt < 2; ++nt)
            acc[mt][nt] = (v8f)0.0f;

    // issue async staging for stage s into buffer s&1
    auto stage = [&](int s) {
        const int kk = s * TK;
        const uint16_t* src = (kk < DM) ? Xsrc : Hsrc;
        const int kb = (kk < DM) ? kk : (kk - DM);
        uint16_t* xs = XsB[s & 1];
        uint16_t* ws = WsB[s & 1];
#pragma unroll
        for (int p = 0; p < 2; ++p) {
            int row = p * 64 + (tid >> 2);
            copy16_to_lds(src + (size_t)(r0 + row) * DM + kb + ch * 8,
                          xs + row * XSTRIDE + ch * 8);
        }
        {
            int row = tid >> 2;                       // 0..63
            const uint16_t* wp = Wl + (size_t)(c0 + row) * KTOT + kk + ch * 8;
            copy16_to_lds(wp, ws + row * WSTRIDE + ch * 8);
            if (kk + 2 * TK < KTOT)                   // warm WGP$ two stages out
                __builtin_prefetch(wp + 2 * TK, 0, 0);
        }
    };

    stage(0);   // prologue prefetch into buffer 0

#pragma unroll
    for (int s = 0; s < NSTAGE; ++s) {
        wait_async0();          // stage s fully landed in LDS (this wave's ops)
        __syncthreads();        // ...and everyone else's; all done reading s-1
        if (s + 1 < NSTAGE) stage(s + 1);   // overlap next stage with compute

        const uint16_t* xs = XsB[s & 1];
        const uint16_t* ws = WsB[s & 1];

        // per-lane fragments (ISA 7.12.2 bf16 layouts)
        union { uint4 u[2]; v16bf v; } fa[2], fb[2];
#pragma unroll
        for (int mt = 0; mt < 2; ++mt) {
            const uint16_t* base = xs + (wr * 32 + mt * 16 + mn) * XSTRIDE;
            fa[mt].u[0] = *(const uint4*)(base + hf * 8);       // K = hf*8..+7
            fa[mt].u[1] = *(const uint4*)(base + 16 + hf * 8);  // K = 16+hf*8..+7
        }
#pragma unroll
        for (int nt = 0; nt < 2; ++nt) {
            const uint16_t* base = ws + (wc * 32 + nt * 16 + mn) * WSTRIDE;
            fb[nt].u[0] = *(const uint4*)(base + hf * 8);       // K = hf*8..+7
            fb[nt].u[1] = *(const uint4*)(base + 16 + hf * 8);  // K = 16+hf*8..+7
        }
#pragma unroll
        for (int mt = 0; mt < 2; ++mt)
#pragma unroll
            for (int nt = 0; nt < 2; ++nt)
                acc[mt][nt] = __builtin_amdgcn_wmma_f32_16x16x32_bf16(
                    false, fa[mt].v, false, fb[nt].v,
                    (short)0, acc[mt][nt], false, false);
    }

    __syncthreads();   // staging buffers dead; smem is reused as Gs below

    // ---- accumulators (+bias) -> LDS gates tile ----
#pragma unroll
    for (int mt = 0; mt < 2; ++mt)
#pragma unroll
        for (int nt = 0; nt < 2; ++nt) {
            int colL = wc * 32 + nt * 16 + mn;        // N = mn
            float bias = bl[c0 + colL];
#pragma unroll
            for (int i = 0; i < 8; ++i) {
                int rowL = wr * 32 + mt * 16 + hf * 8 + i;   // M = hf*8+i
                Gs[rowL * GSTRIDE + colL] = acc[mt][nt][i] + bias;
            }
        }
    __syncthreads();

    // ---- LSTM nonlinearity on this block's (128 rows x 16 d) slice ----
    const int d0 = c0 >> 2;                            // global hidden-unit base
#pragma unroll
    for (int j = 0; j < 8; ++j) {
        int idx = j * 256 + tid;                       // 0..2047, coalesced
        int dl  = idx & 15;
        int row = idx >> 4;
        float4 g4 = *(const float4*)(&Gs[row * GSTRIDE + dl * 4]);  // i,f,g,o
        size_t ci = (size_t)(r0 + row) * DM + d0 + dl;
        float cold = Cst[ci];
        float ig = sigm(g4.x);
        float fg = sigm(g4.y);
        float gg = tanhf(g4.z);
        float og = sigm(g4.w);
        float cn = fg * cold + ig * gg;
        float hn = og * tanhf(cn);
        Cst[ci]  = cn;
        Hdst[ci] = f2bf(hn);
        if (Out)
            Out[(size_t)(r0 + row) * (TAU * DM) + (size_t)t * DM + d0 + dl] = hn;
    }
}

// ---------------- host driver ----------------
extern "C" void kernel_launch(void* const* d_in, const int* in_sizes, int n_in,
                              void* d_out, int out_size, void* d_ws, size_t ws_size,
                              hipStream_t stream)
{
    const float* hidden = (const float*)d_in[0];  // [NL,B,DM]
    const float* cell   = (const float*)d_in[1];  // [NL,B,DM]
    const float* W_ih   = (const float*)d_in[2];  // [NL,4D,D]
    const float* W_hh   = (const float*)d_in[3];  // [NL,4D,D]
    const float* b_ih   = (const float*)d_in[4];  // [NL,4D]
    const float* b_hh   = (const float*)d_in[5];  // [NL,4D]
    float* out = (float*)d_out;                   // [B,TAU,DM]

    // workspace layout (bytes)
    char* ws = (char*)d_ws;
    const size_t W_BYTES = (size_t)NL * GATES * KTOT * 2;           // 3 MiB
    const size_t B_BYTES = 16384;                                   // bias (padded)
    const size_t H_BYTES = (size_t)2 * NL * BATCH * DM * 2;         // ping-pong h
    uint16_t* Wperm = (uint16_t*)ws;
    float*    bsum  = (float*)(ws + W_BYTES);
    uint16_t* hbuf  = (uint16_t*)(ws + W_BYTES + B_BYTES);
    float*    cws   = (float*)(ws + W_BYTES + B_BYTES + H_BYTES);

    // one-time prep (re-run every call: deterministic, inputs untouched)
    prep_weights<<< (NL * GATES * KTOT) / 256, 256, 0, stream>>>(W_ih, W_hh, Wperm);
    prep_bias   <<< (NL * GATES) / 256,        256, 0, stream>>>(b_ih, b_hh, bsum);
    prep_state  <<< (NL * BATCH * DM) / 256,   256, 0, stream>>>(hidden, cell, hbuf, cws);

    const size_t LH = (size_t)BATCH * DM;     // elements per layer h
    dim3 grid(GATES / TN, BATCH / TM);        // (16, 8) = 128 blocks
    for (int t = 0; t < TAU; ++t) {
        int p = t & 1;
        for (int l = 0; l < NL; ++l) {
            const uint16_t* Xsrc = (l == 0)
                ? hbuf + ((size_t)p * NL + (NL - 1)) * LH         // top-layer h, prev step
                : hbuf + ((size_t)(1 - p) * NL + (l - 1)) * LH;   // layer below, this step
            const uint16_t* Hsrc = hbuf + ((size_t)p * NL + l) * LH;
            uint16_t*       Hdst = hbuf + ((size_t)(1 - p) * NL + l) * LH;
            float*          Cl   = cws + (size_t)l * LH;
            float*          Op   = (l == NL - 1) ? out : nullptr;
            lstm_cell_kernel<<<grid, 256, 0, stream>>>(
                Wperm + (size_t)l * GATES * KTOT, bsum + (size_t)l * GATES,
                Xsrc, Hsrc, Hdst, Cl, Op, t);
        }
    }
}
